// _NonLocalBlockND_Group_10273561772700
// MI455X (gfx1250) — matrix operand order repaired and checked by reading, not compiled, not run on hardware
//
#include <hip/hip_runtime.h>
#include <hip/hip_bf16.h>
#include <cstdint>
#include <cstddef>

// ---------------------------------------------------------------------------
// CDNA5 / gfx1250 wave32 WMMA implementation of grouped non-local block + FFN
// ---------------------------------------------------------------------------
typedef __attribute__((ext_vector_type(16))) _Float16 v16h;
typedef __attribute__((ext_vector_type(8)))  float    v8f;

#define N_PIX 3136   // 56*56
#define IMG_W 56
#define BATCH 4
#define EPSF  1e-5f

// ----- WMMA operand layout helpers (16-bit, wave32; this code's contiguous k order, same for A and B) -------
// A (16x32): lane L holds row M=L&15; half h -> K = h + 16*(L>>4)
__device__ __forceinline__ int a_koff(int h, int lane) {
  return h + ((lane >> 4) << 4);
}
// B (32x16): lane L holds col N=L&15; half h -> K = h + 16*(L>>4)
__device__ __forceinline__ int b_koff(int h, int lane) {
  return h + ((lane >> 4) << 4);
}

// Load A tile from row-major f16 matrix p[row][k], leading dim ld
__device__ __forceinline__ v16h load_a_f16(const _Float16* p, int row0, long ld,
                                           int k0, int lane) {
  v16h a;
  const _Float16* rp = p + (long)(row0 + (lane & 15)) * ld + k0;
#pragma unroll
  for (int h = 0; h < 16; ++h) a[h] = rp[a_koff(h, lane)];
  return a;
}

// Load B tile from f16 matrix p[k][col], leading dim ld
__device__ __forceinline__ v16h load_b_f16(const _Float16* p, int k0, long ld,
                                           int col0, int lane) {
  v16h b;
  const _Float16* cp = p + (long)k0 * ld + col0 + (lane & 15);
#pragma unroll
  for (int h = 0; h < 16; ++h) b[h] = cp[(long)b_koff(h, lane) * ld];
  return b;
}

// Load B tile from f32 matrix (convert to f16)
__device__ __forceinline__ v16h load_b_f32(const float* p, int k0, long ld,
                                           int col0, int lane) {
  v16h b;
  const float* cp = p + (long)k0 * ld + col0 + (lane & 15);
#pragma unroll
  for (int h = 0; h < 16; ++h) b[h] = (_Float16)cp[(long)b_koff(h, lane) * ld];
  return b;
}

__device__ __forceinline__ v8f wmma16(v16h a, v16h b, v8f c) {
  // v_wmma_f32_16x16x32_f16 (codegen-confirmed builtin, 8 args)
  return __builtin_amdgcn_wmma_f32_16x16x32_f16(false, a, false, b, (short)0, c,
                                                false, false);
}

// ---------------------------------------------------------------------------
// K0: weight pack + BN fold  (f32 -> f16 layouts, folded scale/shift)
// ---------------------------------------------------------------------------
__global__ void k_pack(const float* g_w, const float* g_b, const float* th_w,
                       const float* th_b, const float* ph_w, const float* ph_b,
                       const float* W_w, const float* W_b, const float* bnWs,
                       const float* bnWb, const float* bnWm, const float* bnWv,
                       const float* f1w, const float* bn1s, const float* bn1b,
                       const float* bn1m, const float* bn1v, const float* f2w,
                       const float* bn2s, const float* bn2b, const float* bn2m,
                       const float* bn2v, const float* f3w, const float* bn3s,
                       const float* bn3b, const float* bn3m, const float* bn3v,
                       _Float16* wf, float* biasf, _Float16* ww16,
                       _Float16* w1h, _Float16* w2h, _Float16* w3h, float* aW,
                       float* bW, float* a1, float* b1, float* a2, float* b2,
                       float* a3, float* b3) {
  int i = blockIdx.x * blockDim.x + threadIdx.x;
  const int R0 = 98304;          // wf  384x256
  const int R1 = R0 + 384;       // biasf
  const int R2 = R1 + 32768;     // ww16 256x128
  const int R3 = R2 + 16384;     // w1h 64x256
  const int R4 = R3 + 36864;     // w2h 64x576
  const int R5 = R4 + 16384;     // w3h 256x64
  const int R6 = R5 + 256;       // aW/bW
  const int R7 = R6 + 64;        // a1/b1
  const int R8 = R7 + 64;        // a2/b2
  const int R9 = R8 + 256;       // a3/b3
  if (i >= R9) return;
  if (i < R0) {
    int o = i >> 8, c = i & 255;
    float v = (o < 128)   ? g_w[o * 256 + c]
              : (o < 256) ? th_w[(o - 128) * 256 + c]
                          : ph_w[(o - 256) * 256 + c];
    wf[i] = (_Float16)v;
  } else if (i < R1) {
    int j = i - R0;
    biasf[j] = (j < 128) ? g_b[j] : (j < 256) ? th_b[j - 128] : ph_b[j - 256];
  } else if (i < R2) {
    ww16[i - R1] = (_Float16)W_w[i - R1];
  } else if (i < R3) {
    w1h[i - R2] = (_Float16)f1w[i - R2];
  } else if (i < R4) {
    int j = i - R3;
    int o = j / 576, k = j - o * 576;
    int ic = k / 9, r = k - ic * 9;  // K index = ic*9 + (kh*3+kw)
    w2h[j] = (_Float16)f2w[(o * 64 + ic) * 9 + r];
  } else if (i < R5) {
    w3h[i - R4] = (_Float16)f3w[i - R4];
  } else if (i < R6) {
    int j = i - R5;
    float inv = bnWs[j] * rsqrtf(bnWv[j] + EPSF);
    aW[j] = inv;
    bW[j] = W_b[j] * inv + bnWb[j] - bnWm[j] * inv;
  } else if (i < R7) {
    int j = i - R6;
    float inv = bn1s[j] * rsqrtf(bn1v[j] + EPSF);
    a1[j] = inv;
    b1[j] = bn1b[j] - bn1m[j] * inv;
  } else if (i < R8) {
    int j = i - R7;
    float inv = bn2s[j] * rsqrtf(bn2v[j] + EPSF);
    a2[j] = inv;
    b2[j] = bn2b[j] - bn2m[j] * inv;
  } else {
    int j = i - R8;
    float inv = bn3s[j] * rsqrtf(bn3v[j] + EPSF);
    a3[j] = inv;
    b3[j] = bn3b[j] - bn3m[j] * inv;
  }
}

// ---------------------------------------------------------------------------
// K1: fused QKV projection  out[384,N] = Wf[384,256] x x[256,N]  (per batch)
//     scatter to attention-friendly f16 layouts.
// ---------------------------------------------------------------------------
__global__ void k_qkv(const float* __restrict__ x, const _Float16* __restrict__ wf,
                      const float* __restrict__ biasf, _Float16* __restrict__ g_t,
                      _Float16* __restrict__ th_t, _Float16* __restrict__ phi) {
  int wave = (int)((blockIdx.x * blockDim.x + threadIdx.x) >> 5);
  int lane = threadIdx.x & 31;
  int nt = wave % 196;
  int rem = wave / 196;
  int mt = rem % 24, b = rem / 24;
  const float* xb = x + (long)b * 256 * N_PIX;
  v8f acc = {};
#pragma unroll
  for (int ks = 0; ks < 256; ks += 32) {
    v16h a = load_a_f16(wf, mt * 16, 256, ks, lane);
    v16h bb = load_b_f32(xb, ks, N_PIX, nt * 16, lane);
    acc = wmma16(a, bb, acc);
  }
  int n = nt * 16 + (lane & 15);
#pragma unroll
  for (int r = 0; r < 8; ++r) {
    int o = mt * 16 + r + ((lane >> 4) << 3);
    _Float16 hv = (_Float16)(acc[r] + biasf[o]);
    int oc = o & 127;
    int grp = oc >> 5, d = oc & 31;
    long bg = (long)b * 4 + grp;
    if (o < 128)
      g_t[(bg * N_PIX + n) * 32 + d] = hv;          // (B,G,N,D)
    else if (o < 256)
      th_t[(bg * N_PIX + n) * 32 + d] = hv;         // (B,G,N,D)
    else
      phi[(bg * 32 + d) * N_PIX + n] = hv;          // (B,G,D,N)
  }
}

// ---------------------------------------------------------------------------
// K2: flash-style grouped attention.  One wave = 16 query rows of one (b,g).
//     S = theta^T phi (2 WMMA / 32 keys), online softmax, O += P x g^T.
// ---------------------------------------------------------------------------
__global__ void __launch_bounds__(256) k_attn(const _Float16* __restrict__ th_t,
                                              const _Float16* __restrict__ phi,
                                              const _Float16* __restrict__ g_t,
                                              _Float16* __restrict__ y) {
  __shared__ _Float16 pbuf[8][16 * 32];  // per-wave P relayout buffer
  int wib = threadIdx.x >> 5;
  int lane = threadIdx.x & 31;
  int wave = blockIdx.x * 8 + wib;
  int qt = wave % 196;
  int bg = wave / 196;  // 0..15
  int b = bg >> 2, grp = bg & 3;
  const _Float16* thp = th_t + (long)bg * N_PIX * 32;
  const _Float16* php = phi + (long)bg * 32 * N_PIX;
  const _Float16* gp = g_t + (long)bg * N_PIX * 32;
  _Float16* pl = pbuf[wib];

  v16h aq = load_a_f16(thp, qt * 16, 32, 0, lane);  // queries, resident
  v8f o0 = {}, o1 = {};
  float mrun[8], lrun[8];
#pragma unroll
  for (int r = 0; r < 8; ++r) {
    mrun[r] = -1e30f;
    lrun[r] = 0.f;
  }

  for (int ms = 0; ms < N_PIX; ms += 32) {
    if (ms + 32 < N_PIX) {  // uniform branch: prefetch next key tiles
      __builtin_prefetch(php + (ms + 32) + (long)(lane & 15) * N_PIX, 0, 1);
      __builtin_prefetch(gp + (long)(ms + 32) * 32, 0, 1);
    }
    v16h bp0 = load_b_f16(php, 0, N_PIX, ms, lane);
    v16h bp1 = load_b_f16(php, 0, N_PIX, ms + 16, lane);
    v8f zc = {};
    v8f s0 = wmma16(aq, bp0, zc);
    v8f s1 = wmma16(aq, bp1, zc);
#pragma unroll
    for (int r = 0; r < 8; ++r) {
      float t = fmaxf(s0[r], s1[r]);
#pragma unroll
      for (int m = 1; m <= 8; m <<= 1) t = fmaxf(t, __shfl_xor(t, m, 32));
      float mn = fmaxf(mrun[r], t);
      float sc = __expf(mrun[r] - mn);
      float p0 = __expf(s0[r] - mn);
      float p1 = __expf(s1[r] - mn);
      float rs = p0 + p1;
#pragma unroll
      for (int m = 1; m <= 8; m <<= 1) rs += __shfl_xor(rs, m, 32);
      lrun[r] = lrun[r] * sc + rs;
      mrun[r] = mn;
      o0[r] *= sc;
      o1[r] *= sc;
      int mrow = r + ((lane >> 4) << 3);  // C-layout row
      pl[mrow * 32 + (lane & 15)] = (_Float16)p0;
      pl[mrow * 32 + 16 + (lane & 15)] = (_Float16)p1;
    }
    asm volatile("s_wait_dscnt 0" ::: "memory");  // wave-local LDS RAW fence
    v16h ap = load_a_f16(pl, 0, 32, 0, lane);     // P as A-operand
    v16h bg0 = load_b_f16(gp, ms, 32, 0, lane);   // g^T tiles (K=m, col=d)
    v16h bg1 = load_b_f16(gp, ms, 32, 16, lane);
    o0 = wmma16(ap, bg0, o0);
    o1 = wmma16(ap, bg1, o1);
  }

  _Float16* yb = y + (long)b * 128 * N_PIX;  // (B,IC,N) f16
#pragma unroll
  for (int r = 0; r < 8; ++r) {
    int n = qt * 16 + r + ((lane >> 4) << 3);
    float inv = 1.0f / lrun[r];
    int d0 = lane & 15;
    yb[(long)(grp * 32 + d0) * N_PIX + n] = (_Float16)(o0[r] * inv);
    yb[(long)(grp * 32 + 16 + d0) * N_PIX + n] = (_Float16)(o1[r] * inv);
  }
}

// ---------------------------------------------------------------------------
// K3: z = relu(BN(W_w x y + W_b) + x); store f32 (residual) + f16 (next GEMM)
// ---------------------------------------------------------------------------
__global__ void k_wz(const _Float16* __restrict__ yh, const _Float16* __restrict__ ww,
                     const float* __restrict__ aW, const float* __restrict__ bW,
                     const float* __restrict__ x, float* __restrict__ z,
                     _Float16* __restrict__ zh) {
  int wave = (int)((blockIdx.x * blockDim.x + threadIdx.x) >> 5);
  int lane = threadIdx.x & 31;
  int nt = wave % 196;
  int rem = wave / 196;
  int mt = rem % 16, b = rem / 16;
  const _Float16* yb = yh + (long)b * 128 * N_PIX;
  v8f acc = {};
#pragma unroll
  for (int ks = 0; ks < 128; ks += 32) {
    v16h a = load_a_f16(ww, mt * 16, 128, ks, lane);
    v16h bb = load_b_f16(yb, ks, N_PIX, nt * 16, lane);
    acc = wmma16(a, bb, acc);
  }
  int n = nt * 16 + (lane & 15);
#pragma unroll
  for (int r = 0; r < 8; ++r) {
    int o = mt * 16 + r + ((lane >> 4) << 3);
    long idx = ((long)b * 256 + o) * N_PIX + n;
    float v = fmaxf(acc[r] * aW[o] + bW[o] + x[idx], 0.f);
    z[idx] = v;
    zh[idx] = (_Float16)v;
  }
}

// ---------------------------------------------------------------------------
// K4: o1 = relu(BN(ff1_w x z))   (64 x 256 GEMM)
// ---------------------------------------------------------------------------
__global__ void k_ff1(const _Float16* __restrict__ zh, const _Float16* __restrict__ w1,
                      const float* __restrict__ a1, const float* __restrict__ b1,
                      _Float16* __restrict__ o1h) {
  int wave = (int)((blockIdx.x * blockDim.x + threadIdx.x) >> 5);
  int lane = threadIdx.x & 31;
  int nt = wave % 196;
  int rem = wave / 196;
  int mt = rem % 4, b = rem / 4;
  const _Float16* zb = zh + (long)b * 256 * N_PIX;
  v8f acc = {};
#pragma unroll
  for (int ks = 0; ks < 256; ks += 32) {
    v16h a = load_a_f16(w1, mt * 16, 256, ks, lane);
    v16h bb = load_b_f16(zb, ks, N_PIX, nt * 16, lane);
    acc = wmma16(a, bb, acc);
  }
  int n = nt * 16 + (lane & 15);
#pragma unroll
  for (int r = 0; r < 8; ++r) {
    int o = mt * 16 + r + ((lane >> 4) << 3);
    float v = fmaxf(acc[r] * a1[o] + b1[o], 0.f);
    o1h[((long)b * 64 + o) * N_PIX + n] = (_Float16)v;
  }
}

// ---------------------------------------------------------------------------
// K5: o2 = relu(BN(conv3x3(o1)))  implicit-GEMM WMMA, K = 64*9 = 576
// ---------------------------------------------------------------------------
__global__ void k_ff2(const _Float16* __restrict__ o1h, const _Float16* __restrict__ w2,
                      const float* __restrict__ a2, const float* __restrict__ b2,
                      _Float16* __restrict__ o2h) {
  int wave = (int)((blockIdx.x * blockDim.x + threadIdx.x) >> 5);
  int lane = threadIdx.x & 31;
  int nt = wave % 196;
  int rem = wave / 196;
  int mt = rem % 4, b = rem / 4;
  const _Float16* ob = o1h + (long)b * 64 * N_PIX;
  int n = nt * 16 + (lane & 15);
  int py = n / IMG_W, px = n - py * IMG_W;
  v8f acc = {};
  for (int ks = 0; ks < 576; ks += 32) {
    v16h a = load_a_f16(w2, mt * 16, 576, ks, lane);
    v16h bb;
#pragma unroll
    for (int h = 0; h < 16; ++h) {
      int K = ks + b_koff(h, lane);
      int ic = K / 9;
      int kk = K - ic * 9;
      int kh = kk / 3;
      int kw = kk - kh * 3;
      int yy = py + kh - 1, xx = px + kw - 1;
      _Float16 v = (_Float16)0.f;
      if ((unsigned)yy < (unsigned)IMG_W && (unsigned)xx < (unsigned)IMG_W)
        v = ob[(long)ic * N_PIX + yy * IMG_W + xx];
      bb[h] = v;
    }
    acc = wmma16(a, bb, acc);
  }
#pragma unroll
  for (int r = 0; r < 8; ++r) {
    int o = mt * 16 + r + ((lane >> 4) << 3);
    float v = fmaxf(acc[r] * a2[o] + b2[o], 0.f);
    o2h[((long)b * 64 + o) * N_PIX + n] = (_Float16)v;
  }
}

// ---------------------------------------------------------------------------
// K6: out = relu(BN(ff3_w x o2) + z)   (256 x 64 GEMM, f32 output)
// ---------------------------------------------------------------------------
__global__ void k_ff3(const _Float16* __restrict__ o2h, const _Float16* __restrict__ w3,
                      const float* __restrict__ a3, const float* __restrict__ b3,
                      const float* __restrict__ z, float* __restrict__ out) {
  int wave = (int)((blockIdx.x * blockDim.x + threadIdx.x) >> 5);
  int lane = threadIdx.x & 31;
  int nt = wave % 196;
  int rem = wave / 196;
  int mt = rem % 16, b = rem / 16;
  const _Float16* ob = o2h + (long)b * 64 * N_PIX;
  v8f acc = {};
#pragma unroll
  for (int ks = 0; ks < 64; ks += 32) {
    v16h a = load_a_f16(w3, mt * 16, 64, ks, lane);
    v16h bb = load_b_f16(ob, ks, N_PIX, nt * 16, lane);
    acc = wmma16(a, bb, acc);
  }
  int n = nt * 16 + (lane & 15);
#pragma unroll
  for (int r = 0; r < 8; ++r) {
    int o = mt * 16 + r + ((lane >> 4) << 3);
    long idx = ((long)b * 256 + o) * N_PIX + n;
    out[idx] = fmaxf(acc[r] * a3[o] + b3[o] + z[idx], 0.f);
  }
}

// ---------------------------------------------------------------------------
extern "C" void kernel_launch(void* const* d_in, const int* in_sizes, int n_in,
                              void* d_out, int out_size, void* d_ws,
                              size_t ws_size, hipStream_t stream) {
  (void)in_sizes; (void)n_in; (void)out_size; (void)ws_size;
  const float* x = (const float*)d_in[0];
  const float* g_w = (const float*)d_in[1];
  const float* g_b = (const float*)d_in[2];
  const float* th_w = (const float*)d_in[3];
  const float* th_b = (const float*)d_in[4];
  const float* ph_w = (const float*)d_in[5];
  const float* ph_b = (const float*)d_in[6];
  const float* W_w = (const float*)d_in[7];
  const float* W_b = (const float*)d_in[8];
  const float* bnWs = (const float*)d_in[9];
  const float* bnWb = (const float*)d_in[10];
  const float* bnWm = (const float*)d_in[11];
  const float* bnWv = (const float*)d_in[12];
  const float* f1w = (const float*)d_in[13];
  const float* bn1s = (const float*)d_in[14];
  const float* bn1b = (const float*)d_in[15];
  const float* bn1m = (const float*)d_in[16];
  const float* bn1v = (const float*)d_in[17];
  const float* f2w = (const float*)d_in[18];
  const float* bn2s = (const float*)d_in[19];
  const float* bn2b = (const float*)d_in[20];
  const float* bn2m = (const float*)d_in[21];
  const float* bn2v = (const float*)d_in[22];
  const float* f3w = (const float*)d_in[23];
  const float* bn3s = (const float*)d_in[24];
  const float* bn3b = (const float*)d_in[25];
  const float* bn3m = (const float*)d_in[26];
  const float* bn3v = (const float*)d_in[27];
  float* out = (float*)d_out;

  // workspace carve-up (256B aligned)
  char* w = (char*)d_ws;
  auto carve = [&](size_t bytes) -> char* {
    char* p = w;
    w += (bytes + 255) & ~(size_t)255;
    return p;
  };
  const long ICN = 4L * 128 * N_PIX;  // 1,605,632
  _Float16* th_t = (_Float16*)carve(ICN * 2);
  _Float16* g_t = (_Float16*)carve(ICN * 2);
  _Float16* phi = (_Float16*)carve(ICN * 2);
  _Float16* yh = (_Float16*)carve(ICN * 2);
  float* z = (float*)carve(4L * 256 * N_PIX * 4);
  _Float16* zh = (_Float16*)carve(4L * 256 * N_PIX * 2);
  _Float16* o1h = (_Float16*)carve(4L * 64 * N_PIX * 2);
  _Float16* o2h = (_Float16*)carve(4L * 64 * N_PIX * 2);
  _Float16* wf = (_Float16*)carve(98304 * 2);
  float* biasf = (float*)carve(384 * 4);
  _Float16* ww16 = (_Float16*)carve(32768 * 2);
  _Float16* w1h = (_Float16*)carve(16384 * 2);
  _Float16* w2h = (_Float16*)carve(36864 * 2);
  _Float16* w3h = (_Float16*)carve(16384 * 2);
  float* aW = (float*)carve(256 * 4);
  float* bW = (float*)carve(256 * 4);
  float* a1 = (float*)carve(64 * 4);
  float* b1 = (float*)carve(64 * 4);
  float* a2 = (float*)carve(64 * 4);
  float* b2 = (float*)carve(64 * 4);
  float* a3 = (float*)carve(256 * 4);
  float* b3 = (float*)carve(256 * 4);

  k_pack<<<789, 256, 0, stream>>>(g_w, g_b, th_w, th_b, ph_w, ph_b, W_w, W_b,
                                  bnWs, bnWb, bnWm, bnWv, f1w, bn1s, bn1b, bn1m,
                                  bn1v, f2w, bn2s, bn2b, bn2m, bn2v, f3w, bn3s,
                                  bn3b, bn3m, bn3v, wf, biasf, ww16, w1h, w2h,
                                  w3h, aW, bW, a1, b1, a2, b2, a3, b3);
  // K1: waves = 4*24*196 = 18816 -> 4704 blocks x 128
  k_qkv<<<4704, 128, 0, stream>>>(x, wf, biasf, g_t, th_t, phi);
  // K2: waves = 16*196 = 3136 -> 392 blocks x 256
  k_attn<<<392, 256, 0, stream>>>(th_t, phi, g_t, yh);
  // K3: waves = 4*16*196 = 12544 -> 3136 blocks x 128
  k_wz<<<3136, 128, 0, stream>>>(yh, ww16, aW, bW, x, z, zh);
  // K4: waves = 4*4*196 = 3136 -> 784 blocks x 128
  k_ff1<<<784, 128, 0, stream>>>(zh, w1h, a1, b1, o1h);
  // K5: same shape as K4
  k_ff2<<<784, 128, 0, stream>>>(o1h, w2h, a2, b2, o2h);
  // K6: waves = 4*16*196 = 12544 -> 3136 blocks x 128
  k_ff3<<<3136, 128, 0, stream>>>(o2h, w3h, a3, b3, z, out);
}
